// RNNModel_53721450938573
// MI455X (gfx1250) — matrix operand order repaired and checked by reading, not compiled, not run on hardware
//
#include <hip/hip_runtime.h>
#include <hip/hip_fp16.h>

#define B_ 512
#define T_ 512
#define I_ 64
#define H_ 256

typedef __attribute__((ext_vector_type(16))) _Float16 v16h;
typedef __attribute__((ext_vector_type(8)))  _Float16 v8h;
typedef __attribute__((ext_vector_type(8)))  float    v8f;

union AV { v16h v; v8h h[2]; };

__device__ __forceinline__ v8f wmma_f16(v16h a, v16h b, v8f c) {
  // D = A(16x32 f16) * B(32x16 f16) + C(16x16 f32)
  return __builtin_amdgcn_wmma_f32_16x16x32_f16(false, a, false, b, (short)0, c,
                                                false, false);
}

__device__ __forceinline__ float fast_tanh(float x) {
#if defined(__has_builtin)
#if __has_builtin(__builtin_amdgcn_tanhf)
  return __builtin_amdgcn_tanhf(x);   // native v_tanh_f32 on gfx1250
#else
  return tanhf(x);
#endif
#else
  return tanhf(x);
#endif
}

// ---------------- prep: fp32 -> fp16 elementwise ----------------
__global__ __launch_bounds__(256) void cvt_f32_f16(const float* __restrict__ src,
                                                   _Float16* __restrict__ dst, int n) {
  int i = blockIdx.x * 256 + threadIdx.x;
  if (i < n) dst[i] = (_Float16)src[i];
}

// ---------------- prep: pack W[H][K] (row-major, fp32) into WMMA B-fragment layout ----
// frag[kt][nt][lane][e] : B element (K = kt*32 + ((lane&16)>>1) + (e&7) + ((e&8)<<1), N = nt*16 + (lane&15))
__global__ __launch_bounds__(256) void pack_bfrag(const float* __restrict__ W,
                                                  _Float16* __restrict__ frag, int K) {
  int idx = blockIdx.x * 256 + threadIdx.x;
  int total = H_ * K;
  if (idx >= total) return;
  int e    = idx & 15;
  int lane = (idx >> 4) & 31;
  int rest = idx >> 9;
  int nt   = rest & 15;        // H/16 == 16
  int kt   = rest >> 4;
  int j = nt * 16 + (lane & 15);
  int k = kt * 32 + ((lane & 16) >> 1) + (e & 7) + ((e & 8) << 1);
  frag[idx] = (_Float16)W[j * K + k];
}

// ---------------- projection: pre[t][bt][nt][lane][8] = in . W_ih^T + b_ih + b_hh ----
__global__ __launch_bounds__(256) void proj_kernel(
    const _Float16* __restrict__ in, int K,
    const _Float16* __restrict__ wfrag,
    const float* __restrict__ b_ih, const float* __restrict__ b_hh,
    _Float16* __restrict__ pre) {
  const int t    = blockIdx.x;
  const int bt   = blockIdx.y;
  const int wave = threadIdx.x >> 5;   // 0..7
  const int lane = threadIdx.x & 31;
  const int nt0  = wave, nt1 = wave + 8;
  const int KT   = K >> 5;
  const int m    = lane & 15;
  const int sel  = (lane & 16) >> 1;   // 0 or 8 (A-layout half-select)

  float bias0 = b_ih[nt0 * 16 + m] + b_hh[nt0 * 16 + m];
  float bias1 = b_ih[nt1 * 16 + m] + b_hh[nt1 * 16 + m];
  v8f acc0, acc1;
#pragma unroll
  for (int r = 0; r < 8; ++r) { acc0[r] = bias0; acc1[r] = bias1; }

  const _Float16* arow = in + ((size_t)(bt * 16 + m) * T_ + t) * K + sel;
  for (int kt = 0; kt < KT; ++kt) {
    AV a;
    a.h[0] = *(const v8h*)(arow + kt * 32);
    a.h[1] = *(const v8h*)(arow + kt * 32 + 16);
    v16h b0 = *(const v16h*)(wfrag + ((size_t)(kt * 16 + nt0) * 32 + lane) * 16);
    v16h b1 = *(const v16h*)(wfrag + ((size_t)(kt * 16 + nt1) * 32 + lane) * 16);
    acc0 = wmma_f16(a.v, b0, acc0);
    acc1 = wmma_f16(a.v, b1, acc1);
  }

  v8h o0, o1;
#pragma unroll
  for (int r = 0; r < 8; ++r) { o0[r] = (_Float16)acc0[r]; o1[r] = (_Float16)acc1[r]; }
  size_t base = ((size_t)t * (B_ / 16) + bt) * 16;
  *(v8h*)(pre + ((base + nt0) * 32 + lane) * 8) = o0;
  *(v8h*)(pre + ((base + nt1) * 32 + lane) * 8) = o1;
}

// ---------------- recurrent scan: h_t = tanh(pre_t + h_{t-1} W_hh^T) ----------------
#define LDS_STRIDE 264  // 256 + pad, keeps 16B alignment (528B row stride)

__global__ __launch_bounds__(512) void scan_kernel(
    const _Float16* __restrict__ pre,
    const _Float16* __restrict__ whh_frag,
    _Float16* __restrict__ hseq) {
  __shared__ _Float16 lds[2][16 * LDS_STRIDE];
  const int bt    = blockIdx.x;
  const int wave  = threadIdx.x >> 5;  // == nt, 0..15
  const int lane  = threadIdx.x & 31;
  const int m     = lane & 15;
  const int sel   = (lane & 16) >> 1;      // 0 or 8
  const int mbase = (lane & 16) ? 8 : 0;   // D-layout row base

  // W_hh B-fragments resident in VGPRs for the whole time loop (64 VGPRs)
  v16h bfr[8];
#pragma unroll
  for (int kt = 0; kt < 8; ++kt)
    bfr[kt] = *(const v16h*)(whh_frag + ((size_t)(kt * 16 + wave) * 32 + lane) * 16);

  // h_{-1} = 0
  for (int i = threadIdx.x; i < 16 * LDS_STRIDE; i += 512) lds[0][i] = (_Float16)0.f;
  __syncthreads();

  const _Float16* prew = pre + (((size_t)bt * 16 + wave) * 32 + lane) * 8;
  const size_t tstride = (size_t)(B_ / 16) * 16 * 32 * 8;  // halves per t-slice
  v8h pcur = *(const v8h*)(prew);  // t = 0 fragment

  int cur = 0;
  for (int t = 0; t < T_; ++t) {
    v8h pnext = pcur;
    if (t + 1 < T_) pnext = *(const v8h*)(prew + (size_t)(t + 1) * tstride);
    if (t + 8 < T_) __builtin_prefetch(prew + (size_t)(t + 8) * tstride, 0, 0);

    v8f acc;
#pragma unroll
    for (int r = 0; r < 8; ++r) acc[r] = (float)pcur[r];

    // A-fragments of h_{t-1} straight from row-major LDS (two b128 loads / k-tile)
    const _Float16* hrow = &lds[cur][m * LDS_STRIDE + sel];
#pragma unroll
    for (int kt = 0; kt < 8; ++kt) {
      AV a;
      a.h[0] = *(const v8h*)(hrow + kt * 32);
      a.h[1] = *(const v8h*)(hrow + kt * 32 + 16);
      acc = wmma_f16(a.v, bfr[kt], acc);
    }

    // tanh + scatter new h into the other LDS buffer (D layout -> row major)
    _Float16* nxt = &lds[cur ^ 1][0];
    const int coln = wave * 16 + m;
#pragma unroll
    for (int r = 0; r < 8; ++r) {
      float hv = fast_tanh(acc[r]);
      nxt[(mbase + r) * LDS_STRIDE + coln] = (_Float16)hv;
    }
    __syncthreads();  // one barrier per step

    // coalesced copy of the fresh h tile to global (next layer's input / FC)
    {
      const int row   = threadIdx.x >> 5;  // 0..15
      const int chunk = lane;              // 0..31 * 8 halves
      v8h hv = *(const v8h*)&lds[cur ^ 1][row * LDS_STRIDE + chunk * 8];
      *(v8h*)(hseq + ((size_t)(bt * 16 + row) * T_ + t) * H_ + chunk * 8) = hv;
    }
    pcur = pnext;
    cur ^= 1;
  }
}

// ---------------- final FC: out[b] = h[b,T-1,:] . w_fc + b_fc ----------------
__global__ __launch_bounds__(256) void fc_kernel(const _Float16* __restrict__ hseq,
                                                 const float* __restrict__ w_fc,
                                                 const float* __restrict__ b_fc,
                                                 float* __restrict__ out) {
  __shared__ float red[256];
  const int b = blockIdx.x;
  const int j = threadIdx.x;
  float v = (float)hseq[((size_t)b * T_ + (T_ - 1)) * H_ + j] * w_fc[j];
  red[j] = v;
  __syncthreads();
  for (int s = 128; s > 0; s >>= 1) {
    if (j < s) red[j] += red[j + s];
    __syncthreads();
  }
  if (j == 0) out[b] = red[0] + b_fc[0];
}

// ---------------- host ----------------
extern "C" void kernel_launch(void* const* d_in, const int* in_sizes, int n_in,
                              void* d_out, int out_size, void* d_ws, size_t ws_size,
                              hipStream_t stream) {
  (void)in_sizes; (void)n_in; (void)out_size; (void)ws_size;
  const float* x    = (const float*)d_in[0];
  const float* w_ih[3] = {(const float*)d_in[1], (const float*)d_in[5], (const float*)d_in[9]};
  const float* w_hh[3] = {(const float*)d_in[2], (const float*)d_in[6], (const float*)d_in[10]};
  const float* b_ih[3] = {(const float*)d_in[3], (const float*)d_in[7], (const float*)d_in[11]};
  const float* b_hh[3] = {(const float*)d_in[4], (const float*)d_in[8], (const float*)d_in[12]};
  const float* w_fc = (const float*)d_in[13];
  const float* b_fc = (const float*)d_in[14];
  float* out = (float*)d_out;

  char* ws = (char*)d_ws;
  size_t off = 0;
  auto take = [&](size_t bytes) { char* p = ws + off; off += (bytes + 255) & ~(size_t)255; return p; };
  _Float16* xh    = (_Float16*)take((size_t)B_ * T_ * I_ * 2);
  _Float16* hseq  = (_Float16*)take((size_t)B_ * T_ * H_ * 2);
  _Float16* pre   = (_Float16*)take((size_t)B_ * T_ * H_ * 2);
  _Float16* wihf[3], *whhf[3];
  wihf[0] = (_Float16*)take((size_t)H_ * I_ * 2);
  whhf[0] = (_Float16*)take((size_t)H_ * H_ * 2);
  wihf[1] = (_Float16*)take((size_t)H_ * H_ * 2);
  whhf[1] = (_Float16*)take((size_t)H_ * H_ * 2);
  wihf[2] = (_Float16*)take((size_t)H_ * H_ * 2);
  whhf[2] = (_Float16*)take((size_t)H_ * H_ * 2);

  // prep
  {
    int n = B_ * T_ * I_;
    cvt_f32_f16<<<n / 256, 256, 0, stream>>>(x, xh, n);
    pack_bfrag<<<(H_ * I_) / 256, 256, 0, stream>>>(w_ih[0], wihf[0], I_);
    pack_bfrag<<<(H_ * H_) / 256, 256, 0, stream>>>(w_hh[0], whhf[0], H_);
    pack_bfrag<<<(H_ * H_) / 256, 256, 0, stream>>>(w_ih[1], wihf[1], H_);
    pack_bfrag<<<(H_ * H_) / 256, 256, 0, stream>>>(w_hh[1], whhf[1], H_);
    pack_bfrag<<<(H_ * H_) / 256, 256, 0, stream>>>(w_ih[2], wihf[2], H_);
    pack_bfrag<<<(H_ * H_) / 256, 256, 0, stream>>>(w_hh[2], whhf[2], H_);
  }

  // layers
  for (int l = 0; l < 3; ++l) {
    const _Float16* in = (l == 0) ? xh : hseq;
    int K = (l == 0) ? I_ : H_;
    proj_kernel<<<dim3(T_, B_ / 16), 256, 0, stream>>>(in, K, wihf[l], b_ih[l], b_hh[l], pre);
    scan_kernel<<<dim3(B_ / 16), 512, 0, stream>>>(pre, whhf[l], hseq);
  }

  fc_kernel<<<dim3(B_), 256, 0, stream>>>(hseq, w_fc, b_fc, out);
}
